// JointNet_8641474199710
// MI455X (gfx1250) — compile-verified, matrix operand order checked by reading, not run on hardware
//
#include <hip/hip_runtime.h>
#include <hip/hip_bf16.h>

// CDNA5 (gfx1250) wave32 WMMA types
typedef __attribute__((ext_vector_type(16))) __bf16 v16bf;
typedef __attribute__((ext_vector_type(8)))  __bf16 v8bf;
typedef __attribute__((ext_vector_type(8)))  float  v8f;

// ---------------------------------------------------------------------------
// Kernel 1: swizzle row-major f32 W[K][N] into bf16 WMMA B-operand fragments.
// Fragment (kb,nf) covers K [kb*32,+32) x N [nf*16,+16). ISA 7.12.2 (16-bit,
// wave32):
//   lane L<16 : col n = nf*16+L,    elems 0..7 = K+0..7,  8..15 = K+16..23
//   lane L>=16: col n = nf*16+L-16, elems 0..7 = K+8..15, 8..15 = K+24..31
// Layout: frag*512 + lane*16 bf16 -> per-lane 32B = 2x global_load_b128.
// ---------------------------------------------------------------------------
__global__ void swizzle_w_frags(const float* __restrict__ W,
                                __bf16* __restrict__ out,
                                int K, int N) {
  const int nfr  = N >> 4;
  const int frag = blockIdx.x;          // frag = kb*nfr + nf
  const int kb   = frag / nfr;
  const int nf   = frag - kb * nfr;
  const int lane = threadIdx.x;         // 0..31
  const int n    = nf * 16 + (lane & 15);
  const int ks   = kb * 32 + ((lane < 16) ? 0 : 8);
  __bf16* dst = out + ((size_t)frag * 32 + lane) * 16;
#pragma unroll
  for (int e = 0; e < 16; ++e) {
    const int k = ks + ((e < 8) ? e : (e + 8));
    dst[e] = (__bf16)W[(size_t)k * N + n];
  }
}

// ---------------------------------------------------------------------------
// Kernel 2: proj = X @ W  (f32 in, bf16 WMMA, f32 out).  X:[M][512], W frags
// for N=512 (nfr=32).  64 rows/block, 256 threads = 8 waves = 4(M)x2(N).
// ---------------------------------------------------------------------------
__global__ __launch_bounds__(256)
void proj_gemm(const float* __restrict__ X,
               const __bf16* __restrict__ Wfrag,   // nfr = 32
               float* __restrict__ out) {          // [M][512]
  __shared__ __align__(16) __bf16 Xs[64][520];     // pad 8: stride 260 dwords
  const int tid      = threadIdx.x;
  const int row_base = blockIdx.x * 64;

  for (int idx = tid; idx < 64 * 512; idx += 256) {
    const int u = idx >> 9, k = idx & 511;
    Xs[u][k] = (__bf16)X[(size_t)(row_base + u) * 512 + k];
  }
  __syncthreads();

  const int lane  = tid & 31;
  const int wave  = tid >> 5;
  const int wm    = wave & 3;           // 4 M-waves
  const int wn    = wave >> 2;          // 2 N-waves
  const int mrow  = wm * 16 + (lane & 15);
  const int khalf = (lane < 16) ? 0 : 8;
  const int rbase = wm * 16 + ((lane < 16) ? 0 : 8);
  const __bf16* bp_lane = Wfrag + (size_t)lane * 16;   // + frag*512 later
  const __bf16* ap0 = &Xs[mrow][khalf];

  for (int nb = 0; nb < 4; ++nb) {      // 4 x 128-col chunks = 512
    const int col0 = nb * 128 + wn * 64;
    const __bf16* bp = bp_lane + (size_t)(col0 >> 4) * 512;
    v8f acc[4] = {v8f{}, v8f{}, v8f{}, v8f{}};
    for (int kb = 0; kb < 16; ++kb) {   // K = 512 = 16 x 32
      const __bf16* ap = ap0 + kb * 32;
      const v8bf a_lo = *(const v8bf*)ap;          // K ks..ks+7
      const v8bf a_hi = *(const v8bf*)(ap + 16);   // K ks+16..ks+23
      const v16bf a = __builtin_shufflevector(a_lo, a_hi,
          0, 1, 2, 3, 4, 5, 6, 7, 8, 9, 10, 11, 12, 13, 14, 15);
#pragma unroll
      for (int j = 0; j < 4; ++j) {
        const v16bf bm = *(const v16bf*)(bp + ((size_t)kb * 32 + j) * 512);
        acc[j] = __builtin_amdgcn_wmma_f32_16x16x32_bf16(
            false, a, false, bm, (short)0, acc[j], false, false);
      }
    }
    float* op = out + (size_t)(row_base + rbase) * 512 + col0 + (lane & 15);
#pragma unroll
    for (int j = 0; j < 4; ++j)
#pragma unroll
      for (int r = 0; r < 8; ++r)
        op[(size_t)r * 512 + j * 16] = acc[j][r];
  }
}

// ---------------------------------------------------------------------------
// Kernel 3: fused joint.  One block per (b,t).  hidden tile (64 x 512 bf16)
// computed ONCE in LDS, then 8 waves (2 M x 4 N) sweep V=1024.
// Wave tile = 32M x 64N (8 accums): each B fragment feeds 2 WMMAs, so W2
// is read only 2x per block from L2 (W2 bf16 = 1 MB, fully L2-resident).
// out[((b*T+t)*64+u)*1024 + v]
// ---------------------------------------------------------------------------
__global__ __launch_bounds__(256)
void joint_kernel(const float* __restrict__ enc_proj,   // [1024][512]
                  const float* __restrict__ dec_proj,   // [256][512]
                  const float* __restrict__ b1,         // [512]
                  const __bf16* __restrict__ W2frag,    // nfr = 64
                  float* __restrict__ out) {
  __shared__ __align__(16) __bf16 hid[64][520];
  __shared__ float enc_s[512];
  __shared__ float b1_s[512];

  const int bt  = blockIdx.x;          // b*256 + t
  const int b   = bt >> 8;
  const int tid = threadIdx.x;

  for (int k = tid; k < 512; k += 256) {
    enc_s[k] = enc_proj[(size_t)bt * 512 + k];
    b1_s[k]  = b1[k];
  }
  __syncthreads();

  const float* dec_base = dec_proj + (size_t)b * 64 * 512;
  for (int idx = tid; idx < 64 * 512; idx += 256) {
    const int u = idx >> 9, k = idx & 511;
    const float h = tanhf(enc_s[k] + dec_base[(size_t)u * 512 + k] + b1_s[k]);
    hid[u][k] = (__bf16)h;
  }
  __syncthreads();

  const int lane  = tid & 31;
  const int wave  = tid >> 5;
  const int wm    = wave & 1;           // 2 M-waves (32 rows each)
  const int wn    = wave >> 1;          // 4 N-waves (64 cols each)
  const int m0    = wm * 32 + (lane & 15);     // A subtile 0 row
  const int khalf = (lane < 16) ? 0 : 8;
  const int rbase = wm * 32 + ((lane < 16) ? 0 : 8);
  const __bf16* bp_lane = W2frag + (size_t)lane * 16;
  const __bf16* ap0 = &hid[m0][khalf];         // subtile 1 = +16 rows
  float* outb = out + (size_t)bt * 64 * 1024;

  for (int nb = 0; nb < 4; ++nb) {      // 4 x 256-col passes = V=1024
    const int col0 = nb * 256 + wn * 64;
    const __bf16* bp = bp_lane + (size_t)(col0 >> 4) * 512;
    v8f acc0[4] = {v8f{}, v8f{}, v8f{}, v8f{}};
    v8f acc1[4] = {v8f{}, v8f{}, v8f{}, v8f{}};
    for (int kb = 0; kb < 16; ++kb) {   // K = H = 512
      const __bf16* ap = ap0 + kb * 32;
      const v8bf a0_lo = *(const v8bf*)ap;
      const v8bf a0_hi = *(const v8bf*)(ap + 16);
      const v8bf a1_lo = *(const v8bf*)(ap + 16 * 520);
      const v8bf a1_hi = *(const v8bf*)(ap + 16 * 520 + 16);
      const v16bf a0 = __builtin_shufflevector(a0_lo, a0_hi,
          0, 1, 2, 3, 4, 5, 6, 7, 8, 9, 10, 11, 12, 13, 14, 15);
      const v16bf a1 = __builtin_shufflevector(a1_lo, a1_hi,
          0, 1, 2, 3, 4, 5, 6, 7, 8, 9, 10, 11, 12, 13, 14, 15);
#pragma unroll
      for (int j = 0; j < 4; ++j) {
        const v16bf bm = *(const v16bf*)(bp + ((size_t)kb * 64 + j) * 512);
        acc0[j] = __builtin_amdgcn_wmma_f32_16x16x32_bf16(
            false, a0, false, bm, (short)0, acc0[j], false, false);
        acc1[j] = __builtin_amdgcn_wmma_f32_16x16x32_bf16(
            false, a1, false, bm, (short)0, acc1[j], false, false);
      }
    }
    // C layout: VGPR r -> row rbase+r (lanes 0-15) / rbase+8+r (lanes 16-31)
    float* op = outb + (size_t)rbase * 1024 + col0 + (lane & 15);
#pragma unroll
    for (int j = 0; j < 4; ++j)
#pragma unroll
      for (int r = 0; r < 8; ++r) {
        op[(size_t)r * 1024 + j * 16]              = acc0[j][r];
        op[(size_t)(r + 16) * 1024 + j * 16]       = acc1[j][r];
      }
  }
}

// ---------------------------------------------------------------------------
extern "C" void kernel_launch(void* const* d_in, const int* in_sizes, int n_in,
                              void* d_out, int out_size, void* d_ws, size_t ws_size,
                              hipStream_t stream) {
  (void)in_sizes; (void)n_in; (void)out_size; (void)ws_size;
  const float* enc = (const float*)d_in[0];   // (4,256,512)
  const float* dec = (const float*)d_in[1];   // (4,64,512)
  const float* W1  = (const float*)d_in[2];   // (1024,512)
  const float* b1  = (const float*)d_in[3];   // (512)
  const float* W2  = (const float*)d_in[4];   // (512,1024)
  float* out = (float*)d_out;                 // (4,256,64,1024)

  char* ws = (char*)d_ws;
  __bf16* w1f  = (__bf16*)(ws);                     // 1024*512 bf16 = 1 MB
  __bf16* w2f  = (__bf16*)(ws + (1 << 20));         //  512*1024 bf16 = 1 MB
  float*  encp = (float*)(ws + (2 << 20));          // 1024*512 f32 = 2 MB
  float*  decp = (float*)(ws + (4 << 20));          //  256*512 f32 = 0.5 MB
  // W1 dec half starts at kb=16: 16 frag-rows * 32 nf * 512 elems
  const __bf16* w1f_dec = w1f + (size_t)16 * 32 * 512;

  // 1) weight swizzle to bf16 WMMA fragments
  swizzle_w_frags<<<(1024 / 32) * (512 / 16), 32, 0, stream>>>(W1, w1f, 1024, 512);
  swizzle_w_frags<<<(512 / 32) * (1024 / 16), 32, 0, stream>>>(W2, w2f, 512, 1024);
  // 2) projections (enc: 1024 rows, dec: 256 rows)
  proj_gemm<<<16, 256, 0, stream>>>(enc, w1f,     encp);
  proj_gemm<<< 4, 256, 0, stream>>>(dec, w1f_dec, decp);
  // 3) fused tanh + big GEMM
  joint_kernel<<<1024, 256, 0, stream>>>(encp, decp, b1, w2f, out);
}
